// VDPWIModel_37056977830538
// MI455X (gfx1250) — matrix operand order repaired and checked by reading, not compiled, not run on hardware
//
#include <hip/hip_runtime.h>

// ---------------- problem constants ----------------
#define BB   256      // batch
#define SS   48       // sequence length
#define HDV  250      // half hidden
#define DD2  500      // full hidden
#define PLEN 32       // crop size
#define NEGV (-10000.0f)

// ---------------- WMMA vector types ----------------
typedef __attribute__((ext_vector_type(16))) __bf16 v16bf;
typedef __attribute__((ext_vector_type(8)))  __bf16 v8bf;
typedef __attribute__((ext_vector_type(8)))  float  v8f;
typedef __attribute__((ext_vector_type(2)))  float  v2f;

__device__ __forceinline__ float wave_sum32(float v) {
    #pragma unroll
    for (int o = 16; o > 0; o >>= 1) v += __shfl_xor(v, o, 32);
    return v;
}

// =====================================================================
// 1) Row norms for all 5 similarity variants, both sides.
//    norms[variant][side][b][s], variant: 0=full 1=fwd 2=bwd 3=sum 4=tree
// =====================================================================
__global__ void norms_kernel(const float* __restrict__ seq1, const float* __restrict__ seq2,
                             const float* __restrict__ tree1, const float* __restrict__ tree2,
                             float* __restrict__ norms) {
    int wave = (blockIdx.x * blockDim.x + threadIdx.x) >> 5;
    int lane = threadIdx.x & 31;
    int total = 2 * BB * SS;
    if (wave >= total) return;
    int side = wave / (BB * SS);
    int rem  = wave % (BB * SS);
    int b = rem / SS, s = rem % SS;
    const float* seq = (side ? seq2 : seq1) + (size_t)(b * SS + s) * DD2;
    const float* tre = (side ? tree2 : tree1) + (size_t)(b * SS + s) * HDV;
    float sf = 0.f, sb = 0.f, ssum = 0.f, st = 0.f;
    for (int k = lane; k < HDV; k += 32) {
        float f = seq[k], bk = seq[k + HDV];
        sf += f * f; sb += bk * bk;
        float u = f + bk; ssum += u * u;
        float t = tre[k]; st += t * t;
    }
    sf = wave_sum32(sf); sb = wave_sum32(sb);
    ssum = wave_sum32(ssum); st = wave_sum32(st);
    if (lane == 0) {
        int base = side * (BB * SS) + b * SS + s;
        const int VS = 2 * BB * SS;
        norms[0 * VS + base] = sqrtf(sf + sb);
        norms[1 * VS + base] = sqrtf(sf);
        norms[2 * VS + base] = sqrtf(sb);
        norms[3 * VS + base] = sqrtf(ssum);
        norms[4 * VS + base] = sqrtf(st);
    }
}

// =====================================================================
// 2) Similarity dot GEMMs, V_WMMA_F32_16X16X4_F32, templated per variant
//    dots[variant][b][i][j] = sum_k A[i][k]*B[j][k]
// =====================================================================
template <int V>
__global__ void sim_gemm_kernel(const float* __restrict__ seq1, const float* __restrict__ seq2,
                                const float* __restrict__ tree1, const float* __restrict__ tree2,
                                float* __restrict__ dots) {
    constexpr int  K    = (V == 0) ? DD2 : HDV;
    constexpr int  LD   = (V == 4) ? HDV : DD2;
    constexpr int  KOFF = (V == 2) ? HDV : 0;
    constexpr bool SUM2 = (V == 3);
    const float* aP = (V == 4) ? tree1 : seq1;
    const float* bP = (V == 4) ? tree2 : seq2;

    int bidx = blockIdx.x / 9;
    int tile = blockIdx.x % 9;
    int mbase = (tile / 3) * 16;
    int nbase = (tile % 3) * 16;
    int lane = threadIdx.x;
    int l15  = lane & 15;
    int half = lane >> 4;

    const float* A  = aP + (size_t)bidx * SS * LD + (size_t)(mbase + l15) * LD + KOFF;
    const float* Bm = bP + (size_t)bidx * SS * LD + (size_t)(nbase + l15) * LD + KOFF;
    int khalf = half * 2;   // ISA 16x4 f32 A/B layout: lane-half selects K pair

    v8f acc = {};
    constexpr int KM = K & ~3;
    #pragma unroll 2
    for (int k0 = 0; k0 < KM; k0 += 4) {
        int kk = k0 + khalf;
        float a0 = A[kk],  a1 = A[kk + 1];
        float b0 = Bm[kk], b1 = Bm[kk + 1];
        if (SUM2) {
            a0 += A[HDV + kk];  a1 += A[HDV + kk + 1];
            b0 += Bm[HDV + kk]; b1 += Bm[HDV + kk + 1];
        }
        v2f av = {a0, a1};
        v2f bv = {b0, b1};
        acc = __builtin_amdgcn_wmma_f32_16x16x4_f32(false, av, false, bv, (short)0, acc, false, false);
    }
    if constexpr ((K & 3) != 0) {   // tail (K=250): only lane-half 0 carries data
        int kk = KM + khalf;
        float a0 = 0.f, a1 = 0.f, b0 = 0.f, b1 = 0.f;
        if (kk < K) {
            a0 = A[kk]; a1 = A[kk + 1]; b0 = Bm[kk]; b1 = Bm[kk + 1];
            if (SUM2) { a0 += A[HDV + kk]; a1 += A[HDV + kk + 1];
                        b0 += Bm[HDV + kk]; b1 += Bm[HDV + kk + 1]; }
        }
        v2f av = {a0, a1};
        v2f bv = {b0, b1};
        acc = __builtin_amdgcn_wmma_f32_16x16x4_f32(false, av, false, bv, (short)0, acc, false, false);
    }
    float* out = dots + ((size_t)V * BB + bidx) * SS * SS;
    #pragma unroll
    for (int v = 0; v < 8; v++) {
        int row = mbase + v + 8 * half;
        out[(size_t)row * SS + nbase + l15] = acc[v];
    }
}

// =====================================================================
// 3) Greedy select: one workgroup per (batch, selector channel)
// =====================================================================
__global__ void greedy_kernel(const float* __restrict__ dots, const float* __restrict__ norms,
                              const float* __restrict__ pad, unsigned char* __restrict__ sel) {
    int chan = blockIdx.y;
    int b = blockIdx.x;
    __shared__ float mm[SS * SS];
    __shared__ float redV[256];
    __shared__ int   redI[256];
    int tid = threadIdx.x;
    int variant = (chan == 2) ? 4 : 3;
    const int VS = 2 * BB * SS;
    const float* d  = dots + ((size_t)variant * BB + b) * SS * SS;
    const float* na = norms + (size_t)variant * VS + b * SS;
    const float* nb = norms + (size_t)variant * VS + BB * SS + b * SS;
    const float* pd = pad + (size_t)b * SS * SS;
    unsigned char* so = sel + ((size_t)chan * BB + b) * SS * SS;

    for (int idx = tid; idx < SS * SS; idx += blockDim.x) {
        int i = idx / SS, j = idx % SS;
        float dot = d[idx];
        float v = (chan == 0) ? dot : dot / (na[i] * nb[j] + 1e-8f);
        mm[idx] = NEGV * pd[idx] + v;
        so[idx] = 0;
    }
    __syncthreads();

    for (int it = 0; it < SS; it++) {
        float bv = -INFINITY; int bi = 0;
        for (int idx = tid; idx < SS * SS; idx += blockDim.x) {
            float v = mm[idx];
            if (v > bv) { bv = v; bi = idx; }
        }
        redV[tid] = bv; redI[tid] = bi;
        __syncthreads();
        for (int off = 128; off > 0; off >>= 1) {
            if (tid < off) {
                float ov = redV[tid + off]; int oi = redI[tid + off];
                if (ov > redV[tid] || (ov == redV[tid] && oi < redI[tid])) {
                    redV[tid] = ov; redI[tid] = oi;
                }
            }
            __syncthreads();
        }
        float mv = redV[0]; int mi = redI[0];
        __syncthreads();
        if (mv > NEGV * 0.5f) {
            int r = mi / SS, c = mi % SS;
            if (tid == 0) so[mi] = 1;
            if (tid < SS) { mm[r * SS + tid] = NEGV; mm[tid * SS + c] = NEGV; }
        }
        __syncthreads();
    }
}

// =====================================================================
// 4) Assemble 15-channel focused cube, crop 32x32 -> NHWC bf16, Cpad=16
// =====================================================================
__global__ void assemble_kernel(const float* __restrict__ dots, const float* __restrict__ norms,
                                const float* __restrict__ pad, const unsigned char* __restrict__ sel,
                                __bf16* __restrict__ x0) {
    int idx = blockIdx.x * blockDim.x + threadIdx.x;
    int total = BB * PLEN * PLEN;
    if (idx >= total) return;
    int x = idx % PLEN, y = (idx / PLEN) % PLEN, b = idx / (PLEN * PLEN);
    size_t sp = (size_t)b * SS * SS + y * SS + x;
    unsigned char s0 = sel[0 * (size_t)BB * SS * SS + sp];
    unsigned char s1 = sel[1 * (size_t)BB * SS * SS + sp];
    unsigned char s2 = sel[2 * (size_t)BB * SS * SS + sp];
    float maskSeq  = (s0 | s1) ? 1.0f : 0.1f;
    float maskTree = s2 ? 1.0f : 0.1f;
    float p = pad[sp];
    float inv = 1.0f - p;
    const int VS = 2 * BB * SS;
    size_t obase = (size_t)idx * 16;
    #pragma unroll
    for (int c = 0; c < 15; c++) {
        int v = c / 3, kind = c % 3;
        float dot = dots[((size_t)v * BB + b) * SS * SS + y * SS + x];
        float na = norms[(size_t)v * VS + b * SS + y];
        float nb = norms[(size_t)v * VS + BB * SS + b * SS + x];
        float val;
        if (kind == 0)      val = dot;
        else if (kind == 1) val = dot / (na * nb + 1e-8f);
        else                val = sqrtf(fmaxf(na * na + nb * nb - 2.0f * dot, 1e-12f));
        float simm = NEGV * p + val;
        float mask = (c < 12) ? maskSeq : maskTree;   // TREE_SCALE == 1.0
        x0[obase + c] = (__bf16)(mask * simm * inv);
    }
    x0[obase + 15] = (__bf16)0.0f;   // channel pad
}

// =====================================================================
// 5) Weight repack: (O,I,3,3) f32 -> [9][R32][CINP] bf16, zero padded
// =====================================================================
__global__ void repack_w_kernel(const float* __restrict__ w, __bf16* __restrict__ o,
                                int O, int I, int R32, int CINP) {
    int idx = blockIdx.x * blockDim.x + threadIdx.x;
    int total = 9 * R32 * CINP;
    if (idx >= total) return;
    int r = idx / (R32 * CINP);
    int rem = idx % (R32 * CINP);
    int row = rem / CINP, c = rem % CINP;
    float v = 0.0f;
    if (row < O && c < I) v = w[((size_t)row * I + c) * 9 + r];
    o[idx] = (__bf16)v;
}

// =====================================================================
// 6) Implicit-GEMM 3x3 SAME conv, bf16 WMMA 16x16x32, NHWC.
//    256-thread blocks (8 waves). Per tap: 32xCin weight tile is staged
//    into LDS with GLOBAL_LOAD_ASYNC_TO_LDS_B128 (ASYNCcnt) and shared by
//    all 8 waves (A-fragments via ds_load_b128); B-fragments are coalesced
//    global b128 loads. One wave -> 16 pixels x 32 output channels.
// =====================================================================
__device__ __forceinline__ v16bf cat8(v8bf lo, v8bf hi) {
    return __builtin_shufflevector(lo, hi, 0, 1, 2, 3, 4, 5, 6, 7,
                                            8, 9, 10, 11, 12, 13, 14, 15);
}

template <int CINP, int COUT, int COUTP, int H, int W>
__global__ void __launch_bounds__(256)
conv_wmma_kernel(const __bf16* __restrict__ act, const __bf16* __restrict__ wt9,
                 const float* __restrict__ bias, __bf16* __restrict__ out) {
    constexpr int HW   = H * W;
    constexpr int R32  = (COUT + 31) & ~31;    // weight row pad
    constexpr int LDSP = CINP + 8;             // +16B row pad: conflict-free ds reads
    constexpr int C8   = CINP / 8;             // 16B chunks per row
    __shared__ __bf16 smem[32 * LDSP];

    int tid  = threadIdx.x;
    int wave = tid >> 5;
    int lane = tid & 31;
    int half = lane >> 4, l15 = lane & 15;
    int nbase = (blockIdx.x * 8 + wave) * 16;
    int mbase = blockIdx.y * 32;
    int n  = nbase + l15;                      // this lane's output pixel
    int bb = n / HW;
    int rem = n % HW;                          // HW, W are powers of two -> shifts
    int py = rem / W, px = rem % W;

    const __bf16* a0row = smem + (size_t)l15 * LDSP;
    const __bf16* a1row = smem + (size_t)(l15 + 16) * LDSP;

    v8f acc0 = {}, acc1 = {};
    #pragma unroll
    for (int r = 0; r < 9; r++) {
        int dy = r / 3 - 1, dx = r % 3 - 1;
        int yy = py + dy, xx = px + dx;
        bool pv = ((unsigned)yy < (unsigned)H) & ((unsigned)xx < (unsigned)W);
        const __bf16* pix  = act + ((size_t)(bb * HW + yy * W + xx)) * CINP;
        const __bf16* wsrc = wt9 + ((size_t)r * R32 + mbase) * CINP;

        // --- async stage of 32 x CINP weight tile into LDS (ASYNCcnt) ---
        for (int t = tid; t < 32 * C8; t += 256) {
            int row = t / C8, c8 = t % C8;
            unsigned lds_off = (unsigned)((row * LDSP + c8 * 8) * 2);        // byte offset in LDS
            unsigned long long ga = (unsigned long long)(const void*)(wsrc + row * CINP + c8 * 8);
            asm volatile("global_load_async_to_lds_b128 %0, %1, off"
                         :: "v"(lds_off), "v"(ga) : "memory");
        }
        asm volatile("s_wait_asynccnt 0" ::: "memory");
        __syncthreads();

        // --- K loop: A from LDS (ds_load_b128), B from global (b128) ---
        for (int k0 = 0; k0 < CINP; k0 += 32) {
            __builtin_prefetch(pix + k0 + 32, 0, 1);     // global_prefetch_b8
            int ka0 = k0 + half * 8, ka1 = k0 + 16 + half * 8;
            v8bf a0lo = {}, a0hi = {}, a1lo = {}, a1hi = {};
            if (ka0 < CINP) { a0lo = *(const v8bf*)(a0row + ka0); a1lo = *(const v8bf*)(a1row + ka0); }
            if (ka1 < CINP) { a0hi = *(const v8bf*)(a0row + ka1); a1hi = *(const v8bf*)(a1row + ka1); }
            int kb0 = k0 + half * 8, kb1 = kb0 + 16;
            v8bf blo = {}, bhi = {};
            if (pv && kb0 < CINP) blo = *(const v8bf*)(pix + kb0);
            if (pv && kb1 < CINP) bhi = *(const v8bf*)(pix + kb1);
            v16bf bfrag = cat8(blo, bhi);
            acc0 = __builtin_amdgcn_wmma_f32_16x16x32_bf16(false, cat8(a0lo, a0hi), false, bfrag,
                                                           (short)0, acc0, false, false);
            acc1 = __builtin_amdgcn_wmma_f32_16x16x32_bf16(false, cat8(a1lo, a1hi), false, bfrag,
                                                           (short)0, acc1, false, false);
        }
        __syncthreads();   // before next tap overwrites the LDS tile
    }

    // Store: each lane holds 8 consecutive output channels for its pixel -> b128 stores
    __bf16* obase = out + (size_t)n * COUTP;
    #pragma unroll
    for (int t = 0; t < 2; t++) {
        int rowb = mbase + t * 16 + 8 * half;
        v8f& acc = t ? acc1 : acc0;
        v8bf o;
        #pragma unroll
        for (int v = 0; v < 8; v++) {
            int row = rowb + v;
            float bi = 0.0f;
            if (row < COUT) bi = bias[row];
            float val = fmaxf(acc[v] + bi, 0.0f);
            if (row >= COUT) val = 0.0f;       // zero channel padding
            o[v] = (__bf16)val;
        }
        if (rowb < COUTP) *(v8bf*)(obase + rowb) = o;
    }
}

// =====================================================================
// 7) 2x2 maxpool, stride 2, NHWC bf16 (C = padded channels)
// =====================================================================
template <int C, int H, int W>
__global__ void pool_kernel(const __bf16* __restrict__ in, __bf16* __restrict__ out) {
    constexpr int Ho = H / 2, Wo = W / 2;
    size_t total = (size_t)BB * Ho * Wo * C;
    size_t i = (size_t)blockIdx.x * blockDim.x + threadIdx.x;
    if (i >= total) return;
    int c  = (int)(i % C);
    int xo = (int)((i / C) % Wo);
    int yo = (int)((i / ((size_t)C * Wo)) % Ho);
    int bb = (int)(i / ((size_t)C * Wo * Ho));
    const __bf16* p = in + (size_t)bb * H * W * C + c;
    float v00 = (float)p[((2 * yo) * W + 2 * xo) * C];
    float v01 = (float)p[((2 * yo) * W + 2 * xo + 1) * C];
    float v10 = (float)p[((2 * yo + 1) * W + 2 * xo) * C];
    float v11 = (float)p[((2 * yo + 1) * W + 2 * xo + 1) * C];
    out[(((size_t)bb * Ho + yo) * Wo + xo) * C + c] =
        (__bf16)fmaxf(fmaxf(v00, v01), fmaxf(v10, v11));
}

// =====================================================================
// 8) Dense layer: h[o][b] = relu(sum_c W[o][c]*act[b][c] + bias[o])
//    f32 WMMA 16x16x4, M=128, N=256, K=128; act is NHWC [B][128] bf16
// =====================================================================
__global__ void dnn_wmma_kernel(const float* __restrict__ wt, const float* __restrict__ bias,
                                const __bf16* __restrict__ act, float* __restrict__ h) {
    int lane = threadIdx.x;
    int nbase = blockIdx.x * 16, mbase = blockIdx.y * 16;
    int half = lane >> 4, l15 = lane & 15;
    int m = mbase + l15, n = nbase + l15;
    v8f acc = {};
    #pragma unroll 4
    for (int k0 = 0; k0 < 128; k0 += 4) {
        int ka = k0 + half * 2;
        v2f av = { wt[(size_t)m * 128 + ka], wt[(size_t)m * 128 + ka + 1] };
        v2f bv = { (float)act[(size_t)n * 128 + ka], (float)act[(size_t)n * 128 + ka + 1] };
        acc = __builtin_amdgcn_wmma_f32_16x16x4_f32(false, av, false, bv, (short)0, acc, false, false);
    }
    #pragma unroll
    for (int v = 0; v < 8; v++) {
        int row = mbase + v + 8 * half;
        h[(size_t)row * BB + n] = fmaxf(acc[v] + bias[row], 0.0f);
    }
}

// =====================================================================
// 9) Output layer + log_softmax (5 labels, trivial)
// =====================================================================
__global__ void logits_kernel(const float* __restrict__ h, const float* __restrict__ ow,
                              const float* __restrict__ ob, float* __restrict__ out) {
    int b = blockIdx.x * blockDim.x + threadIdx.x;
    if (b >= BB) return;
    float lg[5];
    #pragma unroll
    for (int l = 0; l < 5; l++) {
        float s = ob[l];
        for (int o = 0; o < 128; o++) s += ow[l * 128 + o] * h[(size_t)o * BB + b];
        lg[l] = s;
    }
    float mx = lg[0];
    #pragma unroll
    for (int l = 1; l < 5; l++) mx = fmaxf(mx, lg[l]);
    float se = 0.0f;
    #pragma unroll
    for (int l = 0; l < 5; l++) se += expf(lg[l] - mx);
    float lse = mx + logf(se);
    #pragma unroll
    for (int l = 0; l < 5; l++) out[b * 5 + l] = lg[l] - lse;
}

// =====================================================================
// Launch
// =====================================================================
extern "C" void kernel_launch(void* const* d_in, const int* in_sizes, int n_in,
                              void* d_out, int out_size, void* d_ws, size_t ws_size,
                              hipStream_t stream) {
    const float* seq1   = (const float*)d_in[0];
    const float* seq2   = (const float*)d_in[1];
    const float* tree1  = (const float*)d_in[2];
    const float* tree2  = (const float*)d_in[3];
    const float* padc   = (const float*)d_in[4];
    const float* c1w = (const float*)d_in[5];  const float* c1b = (const float*)d_in[6];
    const float* c2w = (const float*)d_in[7];  const float* c2b = (const float*)d_in[8];
    const float* c3w = (const float*)d_in[9];  const float* c3b = (const float*)d_in[10];
    const float* c4w = (const float*)d_in[11]; const float* c4b = (const float*)d_in[12];
    const float* c5w = (const float*)d_in[13]; const float* c5b = (const float*)d_in[14];
    const float* dnw = (const float*)d_in[15]; const float* dnb = (const float*)d_in[16];
    const float* otw = (const float*)d_in[17]; const float* otb = (const float*)d_in[18];
    float* outp = (float*)d_out;

    // ---- workspace layout ----
    char* ws = (char*)d_ws;
    size_t off = 0;
    auto take = [&](size_t bytes) { char* p = ws + off; off = (off + bytes + 255) & ~(size_t)255; return p; };
    float*         dots  = (float*)take(sizeof(float) * 5ull * BB * SS * SS);
    float*         norms = (float*)take(sizeof(float) * 5ull * 2 * BB * SS);
    unsigned char* sel   = (unsigned char*)take(3ull * BB * SS * SS);
    __bf16*        x0    = (__bf16*)take(sizeof(__bf16) * (size_t)BB * PLEN * PLEN * 16);
    // repacked weights: [9][R32][CINP]
    __bf16* w1 = (__bf16*)take(sizeof(__bf16) * 9ull * 128 * 16);
    __bf16* w2 = (__bf16*)take(sizeof(__bf16) * 9ull * 192 * 128);
    __bf16* w3 = (__bf16*)take(sizeof(__bf16) * 9ull * 192 * 168);
    __bf16* w4 = (__bf16*)take(sizeof(__bf16) * 9ull * 192 * 192);
    __bf16* w5 = (__bf16*)take(sizeof(__bf16) * 9ull * 128 * 192);
    __bf16* pre   = (__bf16*)take(sizeof(__bf16) * (size_t)BB * 32 * 32 * 128);  // shared pre-pool scratch (max)
    __bf16* postA = (__bf16*)take(sizeof(__bf16) * (size_t)BB * 16 * 16 * 128);  // ping
    __bf16* postB = (__bf16*)take(sizeof(__bf16) * (size_t)BB * 8 * 8 * 168);    // pong
    float*  hbuf  = (float*)take(sizeof(float) * 128ull * BB);
    (void)ws_size; (void)in_sizes; (void)n_in; (void)out_size;

    // weight repack (f32 -> bf16, [tap][CoutPad32][CinPad8])
    auto rp = [&](const float* s, __bf16* d, int O, int I, int R32, int CINP) {
        int n = 9 * R32 * CINP;
        repack_w_kernel<<<(n + 255) / 256, 256, 0, stream>>>(s, d, O, I, R32, CINP);
    };
    rp(c1w, w1, 128, 15,  128, 16);
    rp(c2w, w2, 164, 128, 192, 128);
    rp(c3w, w3, 192, 164, 192, 168);
    rp(c4w, w4, 192, 192, 192, 192);
    rp(c5w, w5, 128, 192, 128, 192);

    // norms: one wave per (side,b,s)
    {
        int waves = 2 * BB * SS;
        norms_kernel<<<(waves * 32 + 127) / 128, 128, 0, stream>>>(seq1, seq2, tree1, tree2, norms);
    }
    // similarity GEMMs (f32 WMMA), one instantiation per variant
    sim_gemm_kernel<0><<<BB * 9, 32, 0, stream>>>(seq1, seq2, tree1, tree2, dots);
    sim_gemm_kernel<1><<<BB * 9, 32, 0, stream>>>(seq1, seq2, tree1, tree2, dots);
    sim_gemm_kernel<2><<<BB * 9, 32, 0, stream>>>(seq1, seq2, tree1, tree2, dots);
    sim_gemm_kernel<3><<<BB * 9, 32, 0, stream>>>(seq1, seq2, tree1, tree2, dots);
    sim_gemm_kernel<4><<<BB * 9, 32, 0, stream>>>(seq1, seq2, tree1, tree2, dots);
    // greedy selection
    greedy_kernel<<<dim3(BB, 3), 256, 0, stream>>>(dots, norms, padc, sel);
    // assemble conv input (NHWC, Cpad=16)
    assemble_kernel<<<(BB * PLEN * PLEN + 255) / 256, 256, 0, stream>>>(dots, norms, padc, sel, x0);

    // conv stack (bf16 WMMA implicit GEMM, NHWC, async-LDS weight staging) + pools
    conv_wmma_kernel<16, 128, 128, 32, 32><<<dim3(BB * 32 * 32 / 128, 4), 256, 0, stream>>>(x0, w1, c1b, pre);
    pool_kernel<128, 32, 32><<<(BB * 16 * 16 * 128 + 255) / 256, 256, 0, stream>>>(pre, postA);
    conv_wmma_kernel<128, 164, 168, 16, 16><<<dim3(BB * 16 * 16 / 128, 6), 256, 0, stream>>>(postA, w2, c2b, pre);
    pool_kernel<168, 16, 16><<<(BB * 8 * 8 * 168 + 255) / 256, 256, 0, stream>>>(pre, postB);
    conv_wmma_kernel<168, 192, 192, 8, 8><<<dim3(BB * 8 * 8 / 128, 6), 256, 0, stream>>>(postB, w3, c3b, pre);
    pool_kernel<192, 8, 8><<<(BB * 4 * 4 * 192 + 255) / 256, 256, 0, stream>>>(pre, postA);
    conv_wmma_kernel<192, 192, 192, 4, 4><<<dim3(BB * 4 * 4 / 128, 6), 256, 0, stream>>>(postA, w4, c4b, pre);
    pool_kernel<192, 4, 4><<<(BB * 2 * 2 * 192 + 255) / 256, 256, 0, stream>>>(pre, postB);
    conv_wmma_kernel<192, 128, 128, 2, 2><<<dim3(BB * 2 * 2 / 128, 4), 256, 0, stream>>>(postB, w5, c5b, pre);
    pool_kernel<128, 2, 2><<<(BB * 128 + 255) / 256, 256, 0, stream>>>(pre, postA);

    // dense (f32 WMMA) + output log-softmax
    dnn_wmma_kernel<<<dim3(BB / 16, 128 / 16), 32, 0, stream>>>(dnw, dnb, postA, hbuf);
    logits_kernel<<<1, 256, 0, stream>>>(hbuf, otw, otb, outp);
}
